// coordx_net_74285754352085
// MI455X (gfx1250) — matrix operand order repaired and checked by reading, not compiled, not run on hardware
//
#include <hip/hip_runtime.h>

typedef __attribute__((ext_vector_type(16))) _Float16 v16h;
typedef __attribute__((ext_vector_type(8)))  _Float16 v8h;
typedef __attribute__((ext_vector_type(8)))  float    v8f;
typedef __attribute__((ext_vector_type(2)))  float    v2f;
typedef __attribute__((ext_vector_type(4)))  unsigned int v4u;
typedef __attribute__((ext_vector_type(8)))  int      v8i;
typedef __attribute__((ext_vector_type(4)))  int      v4i;

#define Hn 768
#define Wn 768
#define HID 256

// ws blob layout (bytes)
#define OFF_HX16   0           // 768*256*2 = 393216
#define OFF_HY16   393216      // 393216
#define OFF_BLOB   786432      // Bpack 131072 | wm2p 4096 | bm1p 1024  = 136192
#define BLOB_BYTES 136192

// ------------------------------------------------------------------
// TDM: 1-row tensor load, data_size = 8 bytes.  D# per ISA 08_async_tensor.
// amdgpu-toolchain (clang-23) 6-arg builtin:
//   (v4u g0, v8i g1, v4i g2, v4i g3, v8i g4, i32 cpol)
// ------------------------------------------------------------------
__device__ __forceinline__ void tdm_load_1d(unsigned lds_off, const void* gptr,
                                            unsigned nelem8)
{
    unsigned long long ga = (unsigned long long)(uintptr_t)gptr;
    v4u g0;
    g0[0] = 1u;                                   // count=1, user D#
    g0[1] = lds_off;                              // lds_addr (bytes)
    g0[2] = (unsigned)(ga & 0xFFFFFFFFu);         // global_addr[31:0]
    g0[3] = (unsigned)((ga >> 32) & 0x01FFFFFFu)  // global_addr[56:32]
          | 0x80000000u;                          // type = 2 ("image")
    v8i g1;
    g1[0] = 0x00030000;                           // wg_mask=0, data_size=3 (8B)
    g1[1] = (int)((nelem8 & 0xFFFFu) << 16);      // tensor_dim0[15:0]
    g1[2] = 0x00010000;                           // tensor_dim0 hi=0, tensor_dim1=1
    g1[3] = (int)((nelem8 & 0xFFFFu) << 16);      // tile_dim0
    g1[4] = 1;                                    // tile_dim1=1, tile_dim2=0
    g1[5] = (int)nelem8;                          // tensor_dim0_stride lo
    g1[6] = 0;
    g1[7] = 0;
    v4i z4 = {0, 0, 0, 0};
    v8i z8 = {0, 0, 0, 0, 0, 0, 0, 0};
    __builtin_amdgcn_tensor_load_to_lds(g0, g1, z4, z4, z8, 0);
}

// ------------------------------------------------------------------
// Kernel 1: branch MLP (grid (768,2) x 256) -> f16 hx/hy
// ------------------------------------------------------------------
__global__ __launch_bounds__(256) void branch_kernel(
    const float* __restrict__ x,  const float* __restrict__ y,
    const float* __restrict__ Wbx, const float* __restrict__ bbx,
    const float* __restrict__ Wby, const float* __restrict__ bby,
    const float* __restrict__ Wp1, const float* __restrict__ bp1,
    const float* __restrict__ Wp2, const float* __restrict__ bp2,
    _Float16* __restrict__ hx16, _Float16* __restrict__ hy16)
{
    __shared__ __align__(16) float sh[HID];
    __shared__ __align__(16) float sh2[HID];
    const int n = blockIdx.x;
    const int t = threadIdx.x;
    const bool isY = (blockIdx.y != 0);

    const float coord = isY ? y[n] : x[n];
    const float* Wb = isY ? Wby : Wbx;
    const float* bb = isY ? bby : bbx;

    sh[t] = __sinf(coord * Wb[t] + bb[t]);
    __syncthreads();

    float a = bp1[t];
    {
        const float4* wr = (const float4*)(Wp1 + t * HID);
        const float4* sv = (const float4*)sh;
        #pragma unroll 8
        for (int q = 0; q < HID / 4; ++q) {
            float4 w = wr[q]; float4 s = sv[q];
            a += s.x * w.x + s.y * w.y + s.z * w.z + s.w * w.w;
        }
    }
    sh2[t] = __sinf(a);
    __syncthreads();

    float a0 = bp2[t];
    float a1 = bp2[t + HID];
    {
        const float4* w0 = (const float4*)(Wp2 + t * HID);
        const float4* w1 = (const float4*)(Wp2 + (t + HID) * HID);
        const float4* sv = (const float4*)sh2;
        #pragma unroll 8
        for (int q = 0; q < HID / 4; ++q) {
            float4 s = sv[q]; float4 u = w0[q]; float4 v = w1[q];
            a0 += s.x * u.x + s.y * u.y + s.z * u.z + s.w * u.w;
            a1 += s.x * v.x + s.y * v.y + s.z * v.z + s.w * v.w;
        }
    }
    float o = __sinf(a0) + __sinf(a1);
    (isY ? hy16 : hx16)[n * HID + t] = (_Float16)o;
}

// ------------------------------------------------------------------
// Kernel 2: build weight blob = [Bpack f16 | wm2 padded | bm1 permuted]
// B-fragment layout, fragment f = kk*16+nt:
//   lane L holds column N = nt*16 + (L&15),
//   element e holds K = kk*32 + (L>=16)*16 + e.
// ------------------------------------------------------------------
__global__ __launch_bounds__(256) void pack_kernel(
    const float* __restrict__ Wm1, const float* __restrict__ Wm2,
    const float* __restrict__ bm1, unsigned char* __restrict__ blob)
{
    _Float16* Bpack = (_Float16*)blob;
    float*    wm2p  = (float*)(blob + 131072);
    float*    bm1p  = (float*)(blob + 135168);

    const int gid  = blockIdx.x * 256 + threadIdx.x;   // 0..65535
    const int e    = gid & 15;
    const int lane = (gid >> 4) & 31;
    const int nt   = (gid >> 9) & 15;
    const int kk   = gid >> 13;
    const int nn   = nt * 16 + (lane & 15);
    const int k    = kk * 32 + (lane >> 4) * 16 + e;
    Bpack[gid] = (_Float16)Wm1[nn * HID + k];

    if (blockIdx.x == 0) {
        const int t = threadIdx.x;            // t = hidden index n
        float4 v;
        v.x = Wm2[0 * HID + t];
        v.y = Wm2[1 * HID + t];
        v.z = Wm2[2 * HID + t];
        v.w = 0.0f;
        ((float4*)wm2p)[t] = v;
        bm1p[(t & 15) * 16 + (t >> 4)] = bm1[t];   // [n16][nt]
    }
}

// ------------------------------------------------------------------
// Kernel 3: fused  sigmoid( sin(outer(hx,hy)@Wm1^T+bm1) @ Wm2^T + bm2 )
// Tile: 16 i x 8 j per workgroup; wave w owns column j0+w, 1 M-tile.
// grid (48, 96) x 256 threads, 4 waves/SIMD (256-VGPR budget).
// ------------------------------------------------------------------
__global__ __launch_bounds__(256, 4) void main_kernel(
    const _Float16* __restrict__ hx16, const _Float16* __restrict__ hy16,
    const unsigned char* __restrict__ blob,
    const float* __restrict__ bm2,
    float* __restrict__ out)
{
    __shared__ __align__(32) unsigned char sdata[BLOB_BYTES]; // 133 KB
    __shared__ __align__(16) _Float16 shx[16 * HID];          //   8 KB
    __shared__ __align__(16) _Float16 shy[8 * HID];           //   4 KB

    const int t  = threadIdx.x;
    const int i0 = blockIdx.x * 16;
    const int j0 = blockIdx.y * 8;

    // ---- async staging: 3 TDM ops from wave 0, then barrier ----
    if (t < 32) {
        tdm_load_1d((unsigned)(uintptr_t)(void*)sdata, blob, BLOB_BYTES / 8);
        tdm_load_1d((unsigned)(uintptr_t)(void*)shx, hx16 + i0 * HID,
                    (16 * HID * 2) / 8);
        tdm_load_1d((unsigned)(uintptr_t)(void*)shy, hy16 + j0 * HID,
                    (8 * HID * 2) / 8);
        __builtin_amdgcn_s_wait_tensorcnt(0);
    }
    __syncthreads();

    const _Float16* sB   = (const _Float16*)sdata;
    const float*    swm2 = (const float*)(sdata + 131072);
    const float*    sbm1 = (const float*)(sdata + 135168);   // [n16][nt]

    const int w    = t >> 5;
    const int lane = t & 31;
    const int half = lane >> 4;
    const int n16  = lane & 15;
    const int j    = j0 + w;
    const _Float16* hyrow = shy + w * HID;
    const _Float16* hr    = shx + n16 * HID;    // A row M = n16
    const _Float16* bB    = sB + lane * 16;     // per-lane B base

    // ---- main loop: 8 K-steps x 16 N-tiles, 128 WMMAs.
    //      kk==0 uses inline-0 C (bm1 added in epilogue).
    //      B fragments are software-pipelined (double-buffered pairs).
    v8f acc[16];
    #pragma unroll
    for (int kk = 0; kk < 8; ++kk) {
        const int kb = kk * 32 + half * 16; // A layout: K = kb + {0..15}
        v8h xa = *(const v8h*)(hr + kb);
        v8h xc = *(const v8h*)(hr + kb + 8);
        v8h ya = *(const v8h*)(hyrow + kb);
        v8h yc = *(const v8h*)(hyrow + kb + 8);
        v8h lo = xa * ya;                   // v_pk_mul_f16
        v8h hi = xc * yc;
        v16h a = __builtin_shufflevector(lo, hi,
                 0,1,2,3,4,5,6,7,8,9,10,11,12,13,14,15);

        const _Float16* bk = bB + kk * 8192;     // halves; frag nt at +nt*512
        v16h b0 = *(const v16h*)(bk);
        v16h b1 = *(const v16h*)(bk + 512);
        #pragma unroll
        for (int nt = 0; nt < 16; nt += 2) {
            v16h c0 = b0, c1 = b1;
            if (nt + 2 < 16) {                   // prefetch next pair
                b0 = *(const v16h*)(bk + (nt + 2) * 512);
                b1 = *(const v16h*)(bk + (nt + 3) * 512);
            }
            if (kk == 0) {
                const v8f zc = {};
                acc[nt]     = __builtin_amdgcn_wmma_f32_16x16x32_f16(
                    false, a, false, c0, (short)0, zc, false, false);
                acc[nt + 1] = __builtin_amdgcn_wmma_f32_16x16x32_f16(
                    false, a, false, c1, (short)0, zc, false, false);
            } else {
                acc[nt]     = __builtin_amdgcn_wmma_f32_16x16x32_f16(
                    false, a, false, c0, (short)0, acc[nt], false, false);
                acc[nt + 1] = __builtin_amdgcn_wmma_f32_16x16x32_f16(
                    false, a, false, c1, (short)0, acc[nt + 1], false, false);
            }
        }
    }

    // ---- bias vector: 4 x b128 (bm1 pre-permuted), loaded after main loop ----
    float bb[16];
    {
        const float4* bp = (const float4*)(sbm1 + n16 * 16);
        #pragma unroll
        for (int q = 0; q < 4; ++q) {
            float4 v = bp[q];
            bb[q * 4 + 0] = v.x; bb[q * 4 + 1] = v.y;
            bb[q * 4 + 2] = v.z; bb[q * 4 + 3] = v.w;
        }
    }

    // ---- epilogue: +bm1, sin, 256->3 projection (v_pk_fma_f32), reduce ----
    v2f p2[4][3];
    #pragma unroll
    for (int rp = 0; rp < 4; ++rp)
        #pragma unroll
        for (int c = 0; c < 3; ++c) { p2[rp][c][0] = 0.f; p2[rp][c][1] = 0.f; }

    #pragma unroll
    for (int nt = 0; nt < 16; ++nt) {
        float4 wv = *(const float4*)(swm2 + (nt * 16 + n16) * 4);
        const float bv = bb[nt];
        float s_[8];
        #pragma unroll
        for (int r = 0; r < 8; ++r) s_[r] = __sinf(acc[nt][r] + bv);
        v2f wx = {wv.x, wv.x};
        v2f wy = {wv.y, wv.y};
        v2f wz = {wv.z, wv.z};
        #pragma unroll
        for (int rp = 0; rp < 4; ++rp) {
            v2f h2; h2[0] = s_[2 * rp]; h2[1] = s_[2 * rp + 1];
            p2[rp][0] = __builtin_elementwise_fma(h2, wx, p2[rp][0]);
            p2[rp][1] = __builtin_elementwise_fma(h2, wy, p2[rp][1]);
            p2[rp][2] = __builtin_elementwise_fma(h2, wz, p2[rp][2]);
        }
    }
    #pragma unroll
    for (int rp = 0; rp < 4; ++rp)
        #pragma unroll
        for (int c = 0; c < 3; ++c) {
            float vx = p2[rp][c][0], vy = p2[rp][c][1];
            #pragma unroll
            for (int m = 1; m < 16; m <<= 1) {
                vx += __shfl_xor(vx, m, 16);
                vy += __shfl_xor(vy, m, 16);
            }
            p2[rp][c][0] = vx; p2[rp][c][1] = vy;
        }

    if (n16 < 3) {
        const int c = n16;
        const float b = bm2[c];
        #pragma unroll
        for (int r = 0; r < 8; ++r) {
            const int i = i0 + half * 8 + r;        // D row M = r + 8*half
            float v = p2[r >> 1][c][r & 1] + b;
            out[(c * Wn + j) * Hn + i] = 1.0f / (1.0f + __expf(-v));
        }
    }
}

// ------------------------------------------------------------------
extern "C" void kernel_launch(void* const* d_in, const int* in_sizes, int n_in,
                              void* d_out, int out_size, void* d_ws, size_t ws_size,
                              hipStream_t stream)
{
    (void)in_sizes; (void)n_in; (void)out_size; (void)ws_size;
    const float* x   = (const float*)d_in[0];
    const float* y   = (const float*)d_in[1];
    const float* Wbx = (const float*)d_in[2];
    const float* bbx = (const float*)d_in[3];
    const float* Wby = (const float*)d_in[4];
    const float* bby = (const float*)d_in[5];
    const float* Wp1 = (const float*)d_in[6];
    const float* bp1 = (const float*)d_in[7];
    const float* Wp2 = (const float*)d_in[8];
    const float* bp2 = (const float*)d_in[9];
    const float* Wm1 = (const float*)d_in[10];
    const float* bm1 = (const float*)d_in[11];
    const float* Wm2 = (const float*)d_in[12];
    const float* bm2 = (const float*)d_in[13];

    char* ws = (char*)d_ws;
    _Float16*      hx16 = (_Float16*)(ws + OFF_HX16);
    _Float16*      hy16 = (_Float16*)(ws + OFF_HY16);
    unsigned char* blob = (unsigned char*)(ws + OFF_BLOB);

    branch_kernel<<<dim3(768, 2), 256, 0, stream>>>(
        x, y, Wbx, bbx, Wby, bby, Wp1, bp1, Wp2, bp2, hx16, hy16);
    pack_kernel<<<dim3(256), 256, 0, stream>>>(Wm1, Wm2, bm1, blob);
    main_kernel<<<dim3(48, 96), 256, 0, stream>>>(
        hx16, hy16, blob, bm2, (float*)d_out);
}
